// ResponseAttention_85246510891215
// MI455X (gfx1250) — matrix operand order repaired and checked by reading, not compiled, not run on hardware
//
#include <hip/hip_runtime.h>
#include <hip/hip_bf16.h>

#define DEVINL __device__ __forceinline__

typedef __attribute__((ext_vector_type(16))) __bf16 v16bf;
typedef __attribute__((ext_vector_type(8)))  float  v8f;

union Frag16 { v16bf v; unsigned int u[8]; };

constexpr int Bc  = 4;
constexpr int Sc  = 4096;
constexpr int Hc  = 512;
constexpr int BSc = Bc * Sc;

// float -> bf16 round-to-nearest-even, kept as raw ushort so all storage is plain integer.
DEVINL unsigned short f2bf(float f) {
  unsigned int x = __float_as_uint(f);
  x += 0x7FFFu + ((x >> 16) & 1u);
  return (unsigned short)(x >> 16);
}

// CDNA5 async copy: global -> LDS, 16 bytes per lane, tracked by ASYNCcnt.
// VDST operand carries the LDS byte offset (low 32 bits of the generic pointer).
DEVINL void async_ld16(void* lds_dst, const void* gsrc) {
  unsigned int       loff  = (unsigned int)(uintptr_t)lds_dst;
  unsigned long long gaddr = (unsigned long long)(uintptr_t)gsrc;
  asm volatile("global_load_async_to_lds_b128 %0, %1, off"
               :: "v"(loff), "v"(gaddr) : "memory");
}
DEVINL void wait_async0() {
  asm volatile("s_wait_asynccnt 0" ::: "memory");
}

// A-matrix 16x32 bf16 fragment (ISA 7.12.2): lane<16 -> M=lane, K-halves {0,16};
// lane>=16 -> M=lane-16, K offset +8 within each half. Pairs are contiguous 32-bit reads.
DEVINL Frag16 load_a_frag(const unsigned short* base, int ld, int kk, int lane) {
  Frag16 f;
  const unsigned short* row = base + (lane & 15) * ld + kk + ((lane >> 4) << 3);
  #pragma unroll
  for (int v = 0; v < 4; ++v) f.u[v]     = *(const unsigned int*)(row + 2 * v);
  #pragma unroll
  for (int v = 0; v < 4; ++v) f.u[4 + v] = *(const unsigned int*)(row + 16 + 2 * v);
  return f;
}

// B-matrix 32x16 bf16 fragment from an [N][K]-major tile: lane<16 -> K-halves {0,16},
// lane>=16 -> K offset +8 within each half (same order as A). Contiguous 32-bit reads.
DEVINL Frag16 load_b_frag(const unsigned short* base, int ld, int kk, int lane) {
  Frag16 f;
  const unsigned short* row = base + (lane & 15) * ld + kk + ((lane >> 4) << 3);
  #pragma unroll
  for (int v = 0; v < 4; ++v) f.u[v]     = *(const unsigned int*)(row + 2 * v);
  #pragma unroll
  for (int v = 0; v < 4; ++v) f.u[4 + v] = *(const unsigned int*)(row + 16 + 2 * v);
  return f;
}

DEVINL v8f wmma_bf16(const Frag16& a, const Frag16& b, v8f c) {
  return __builtin_amdgcn_wmma_f32_16x16x32_bf16(
      /*neg_a=*/false, a.v, /*neg_b=*/false, b.v,
      /*c_mod=*/(short)0, c, /*reuse_a=*/false, /*reuse_b=*/false);
}

// ---------------------------------------------------------------------------
// Kernel 1: fused projections. proj 0: Q = (x Wq^T + bq)/sqrt(H)   -> Qs[row][d]
//                              proj 1: Ksum = (x+rp) Wk^T + 2 bk   -> Ks[row][d]
//                              proj 2: V = x Wv^T + bv             -> Vt[b][d][s]
// V is stored TRANSPOSED so the attention kernel can stage its V tile with
// plain contiguous async b128 copies (no in-loop transpose).
// ---------------------------------------------------------------------------
__global__ __launch_bounds__(256)
void proj_qkv_kernel(const float* __restrict__ x,  const float* __restrict__ rp,
                     const float* __restrict__ Wq, const float* __restrict__ bq,
                     const float* __restrict__ Wk, const float* __restrict__ bk,
                     const float* __restrict__ Wv, const float* __restrict__ bv,
                     unsigned short* __restrict__ Qs,
                     unsigned short* __restrict__ Ks,
                     unsigned short* __restrict__ Vt) {
  const int proj  = blockIdx.z;
  const int mbase = blockIdx.x * 128;
  const int nbase = blockIdx.y * 128;

  const float* W      = (proj == 0) ? Wq : (proj == 1) ? Wk : Wv;
  const float* bias   = (proj == 0) ? bq : (proj == 1) ? bk : bv;
  unsigned short* Out = (proj == 0) ? Qs : (proj == 1) ? Ks : Vt;

  __shared__ unsigned short As[128][72];  // 128 rows x 64 K (+8 pad) bf16
  __shared__ unsigned short Bs[128][72];  // 128 N-rows x 64 K (+8 pad) bf16 (W is [N][K])

  const int tid  = threadIdx.x;
  const int wave = tid >> 5;
  const int lane = tid & 31;
  const int m0   = (wave & 3) * 32;   // wave's M sub-tile (32 rows)
  const int n0   = (wave >> 2) * 64;  // wave's N sub-tile (64 cols)

  const v8f vzero = {0.f, 0.f, 0.f, 0.f, 0.f, 0.f, 0.f, 0.f};
  v8f acc[2][4];
  #pragma unroll
  for (int i = 0; i < 2; ++i)
    #pragma unroll
    for (int j = 0; j < 4; ++j) acc[i][j] = vzero;

  for (int kk = 0; kk < Hc; kk += 64) {
    // Stage A (x or x+rp), converting f32 -> bf16.
    for (int i = tid; i < 2048; i += 256) {
      int r = i >> 4, c4 = (i & 15) << 2;
      float4 f = *(const float4*)(x + (size_t)(mbase + r) * Hc + kk + c4);
      if (proj == 1) {
        float4 g = *(const float4*)(rp + (size_t)(mbase + r) * Hc + kk + c4);
        f.x += g.x; f.y += g.y; f.z += g.z; f.w += g.w;
      }
      unsigned short* d = &As[r][c4];
      d[0] = f2bf(f.x); d[1] = f2bf(f.y); d[2] = f2bf(f.z); d[3] = f2bf(f.w);
    }
    // Stage B (weight rows nbase..nbase+127, W stored [N=512][K=512] row-major).
    for (int i = tid; i < 2048; i += 256) {
      int r = i >> 4, c4 = (i & 15) << 2;
      float4 f = *(const float4*)(W + (size_t)(nbase + r) * Hc + kk + c4);
      unsigned short* d = &Bs[r][c4];
      d[0] = f2bf(f.x); d[1] = f2bf(f.y); d[2] = f2bf(f.z); d[3] = f2bf(f.w);
    }
    __syncthreads();

    #pragma unroll
    for (int k2 = 0; k2 < 64; k2 += 32) {
      Frag16 a0 = load_a_frag(&As[m0][0],      72, k2, lane);
      Frag16 a1 = load_a_frag(&As[m0 + 16][0], 72, k2, lane);
      #pragma unroll
      for (int j = 0; j < 4; ++j) {
        Frag16 bfr = load_b_frag(&Bs[n0 + 16 * j][0], 72, k2, lane);
        acc[0][j] = wmma_bf16(a0, bfr, acc[0][j]);
        acc[1][j] = wmma_bf16(a1, bfr, acc[1][j]);
      }
    }
    __syncthreads();
  }

  const float scale = (proj == 0) ? 0.04419417382415922f : 1.0f;  // 1/sqrt(512)
  const float bmul  = (proj == 1) ? 2.0f : 1.0f;
  const int   rofs  = (lane >> 4) ? 8 : 0;
  #pragma unroll
  for (int mi = 0; mi < 2; ++mi) {
    #pragma unroll
    for (int j = 0; j < 4; ++j) {
      const int col = nbase + n0 + 16 * j + (lane & 15);
      const float bb = bias[col] * bmul;
      #pragma unroll
      for (int r = 0; r < 8; ++r) {
        const int row = mbase + m0 + 16 * mi + rofs + r;  // global token index
        const unsigned short val = f2bf((acc[mi][j][r] + bb) * scale);
        if (proj == 2) {
          // transposed: Vt[b][d=col][s]
          Out[(((size_t)(row >> 12)) * Hc + col) * Sc + (row & (Sc - 1))] = val;
        } else {
          Out[(size_t)row * Hc + col] = val;
        }
      }
    }
  }
}

// ---------------------------------------------------------------------------
// Kernel 2: flash attention. Block = 8 waves x 16 query rows = 128 queries.
// Wave keeps Q (16x512) in 16 A-frags and the 16x512 f32 output accumulator
// in registers; K and transposed-V tiles are staged into LDS with
// global_load_async_to_lds_b128 (ASYNCcnt) each 64-key block.
// ---------------------------------------------------------------------------
__global__ __launch_bounds__(256, 1)
void flash_attn_kernel(const unsigned short* __restrict__ Qs,
                       const unsigned short* __restrict__ Ks,
                       const unsigned short* __restrict__ Vts,
                       float* __restrict__ out) {
  const int b     = blockIdx.y;
  const int qbase = blockIdx.x * 128;
  const int tid = threadIdx.x, wave = tid >> 5, lane = tid & 31;
  const size_t rowbase = (size_t)b * Sc + qbase + wave * 16;  // wave's first query row

  __shared__ unsigned short Kl[64][520];   // 64 keys x 512 d (+8 pad)      66.5 KB
  __shared__ unsigned short Vl[512][72];   // Vt tile: [d][key] (+8 pad)    73.7 KB
  __shared__ unsigned short Pl[128][72];   // probabilities, wave-private   18.4 KB

  // Load this wave's Q panel (16 rows x 512) as 16 resident A-fragments.
  Frag16 qf[16];
  {
    const unsigned short* qrow =
        Qs + (rowbase + (lane & 15)) * Hc + ((lane >> 4) << 3);
    #pragma unroll
    for (int ks = 0; ks < 16; ++ks) {
      #pragma unroll
      for (int v = 0; v < 4; ++v)
        qf[ks].u[v] = *(const unsigned int*)(qrow + ks * 32 + 2 * v);
      #pragma unroll
      for (int v = 0; v < 4; ++v)
        qf[ks].u[4 + v] = *(const unsigned int*)(qrow + ks * 32 + 16 + 2 * v);
    }
  }

  const v8f vzero = {0.f, 0.f, 0.f, 0.f, 0.f, 0.f, 0.f, 0.f};
  v8f acc[32];                       // 16 rows x 512 d output accumulator
  #pragma unroll
  for (int j = 0; j < 32; ++j) acc[j] = vzero;
  float mrow[8], lrow[8];            // per-lane-half row statistics
  #pragma unroll
  for (int r = 0; r < 8; ++r) { mrow[r] = -3.0e38f; lrow[r] = 0.f; }

  for (int kb = 0; kb < Sc; kb += 64) {
    // Stage K tile (64 x 512 bf16) via async b128 copies: 16 per thread.
    for (int i = tid; i < 4096; i += 256) {
      int r = i >> 6, c8 = (i & 63) << 3;
      async_ld16(&Kl[r][c8], Ks + ((size_t)b * Sc + kb + r) * Hc + c8);
    }
    // Stage transposed V tile (512 x 64 bf16) — contiguous rows of Vts[b][d][*].
    for (int i = tid; i < 4096; i += 256) {
      int d = i >> 3, c8 = (i & 7) << 3;
      async_ld16(&Vl[d][c8], Vts + ((size_t)b * Hc + d) * Sc + kb + c8);
    }
    if (kb + 64 < Sc) {  // prefetch next K tile into cache (global_prefetch_b8)
      __builtin_prefetch(Ks + ((size_t)b * Sc + kb + 64) * Hc + tid * 128, 0, 0);
    }
    wait_async0();       // s_wait_asynccnt 0: this wave's tiles are in LDS
    __syncthreads();     // all waves' tiles visible

    // Scores: S(16x64) = Qscaled(16x512) @ K^T ; K=512 -> 16 WMMA k-steps.
    v8f sc[4];
    #pragma unroll
    for (int j = 0; j < 4; ++j) sc[j] = vzero;
    #pragma unroll
    for (int ks = 0; ks < 16; ++ks) {
      #pragma unroll
      for (int j = 0; j < 4; ++j) {
        Frag16 bf = load_b_frag(&Kl[16 * j][0], 520, ks * 32, lane);
        sc[j] = wmma_bf16(qf[ks], bf, sc[j]);
      }
    }

    // Online softmax. Row r lives in lanes 0-15 (rows 0-7) / 16-31 (rows 8-15);
    // 16-lane xor-shuffles stay within each half.
    #pragma unroll
    for (int r = 0; r < 8; ++r) {
      float mx = fmaxf(fmaxf(sc[0][r], sc[1][r]), fmaxf(sc[2][r], sc[3][r]));
      mx = fmaxf(mx, __shfl_xor(mx, 1));
      mx = fmaxf(mx, __shfl_xor(mx, 2));
      mx = fmaxf(mx, __shfl_xor(mx, 4));
      mx = fmaxf(mx, __shfl_xor(mx, 8));
      const float nm = fmaxf(mrow[r], mx);
      const float fr = __expf(mrow[r] - nm);
      mrow[r] = nm;
      float sum = 0.f;
      #pragma unroll
      for (int j = 0; j < 4; ++j) {
        float p = __expf(sc[j][r] - nm);
        sc[j][r] = p;
        sum += p;
      }
      sum += __shfl_xor(sum, 1);
      sum += __shfl_xor(sum, 2);
      sum += __shfl_xor(sum, 4);
      sum += __shfl_xor(sum, 8);
      lrow[r] = lrow[r] * fr + sum;
      #pragma unroll
      for (int j2 = 0; j2 < 32; ++j2) acc[j2][r] *= fr;
    }

    // C-layout -> A-layout for P via wave-private LDS (DS ops are in-order per wave).
    {
      const int prow = wave * 16 + ((lane >> 4) ? 8 : 0);
      const int pcol = lane & 15;
      #pragma unroll
      for (int r = 0; r < 8; ++r)
        #pragma unroll
        for (int j = 0; j < 4; ++j)
          Pl[prow + r][16 * j + pcol] = f2bf(sc[j][r]);
    }

    // acc += P(16x64) @ V(64x512): 2 k-steps x 32 d-tiles.
    #pragma unroll
    for (int ks2 = 0; ks2 < 2; ++ks2) {
      Frag16 a = load_a_frag(&Pl[wave * 16][0], 72, ks2 * 32, lane);
      #pragma unroll
      for (int j2 = 0; j2 < 32; ++j2) {
        Frag16 bf = load_b_frag(&Vl[16 * j2][0], 72, ks2 * 32, lane);
        acc[j2] = wmma_bf16(a, bf, acc[j2]);
      }
    }
    __syncthreads();  // all reads done before next block's async writes land
  }

  // Normalize and write f32 output.
  {
    const int col0 = lane & 15;
    const int rofs = (lane >> 4) ? 8 : 0;
    #pragma unroll
    for (int r = 0; r < 8; ++r) {
      const float inv = 1.0f / lrow[r];
      float* orow = out + (rowbase + rofs + r) * Hc;
      #pragma unroll
      for (int j2 = 0; j2 < 32; ++j2)
        orow[16 * j2 + col0] = acc[j2][r] * inv;
    }
  }
}

// ---------------------------------------------------------------------------
extern "C" void kernel_launch(void* const* d_in, const int* in_sizes, int n_in,
                              void* d_out, int out_size, void* d_ws, size_t ws_size,
                              hipStream_t stream) {
  const float* x  = (const float*)d_in[0];
  const float* rp = (const float*)d_in[1];
  const float* Wq = (const float*)d_in[2];
  const float* bq = (const float*)d_in[3];
  const float* Wk = (const float*)d_in[4];
  const float* bk = (const float*)d_in[5];
  const float* Wv = (const float*)d_in[6];
  const float* bv = (const float*)d_in[7];

  // Workspace: bf16 Q (pre-scaled) [BS][H], Ksum [BS][H], V transposed [B][H][S].
  unsigned short* Qs = (unsigned short*)d_ws;
  unsigned short* Ks = Qs + (size_t)BSc * Hc;
  unsigned short* Vt = Ks + (size_t)BSc * Hc;

  dim3 gp(BSc / 128, Hc / 128, 3);
  proj_qkv_kernel<<<gp, 256, 0, stream>>>(x, rp, Wq, bq, Wk, bk, Wv, bv,
                                          Qs, Ks, Vt);

  dim3 ga(Sc / 128, Bc);
  flash_attn_kernel<<<ga, 256, 0, stream>>>(Qs, Ks, Vt, (float*)d_out);
}
